// OutputColorModel_82162724372505
// MI455X (gfx1250) — matrix operand order repaired and checked by reading, not compiled, not run on hardware
//
#include <hip/hip_runtime.h>
#include <hip/hip_bf16.h>

// ---------------------------------------------------------------------------
// Types for WMMA bf16 path (gfx1250 / CDNA5, wave32)
// ---------------------------------------------------------------------------
typedef __bf16 bf16_t;
typedef __attribute__((ext_vector_type(16))) __bf16        v16bf;
typedef __attribute__((ext_vector_type(8)))  __bf16        v8bf;
typedef __attribute__((ext_vector_type(8)))  float         v8f;
typedef __attribute__((ext_vector_type(4)))  unsigned int  v4u;
typedef __attribute__((ext_vector_type(4)))  int           v4i_t;

union FragBF16 { v4u u[4]; v16bf v; };   // 64 bytes; v16bf v = 32 bytes = 8 VGPRs
union PackBF8  { v4u u;    v8bf  v; };   // 16 bytes

// Model geometry
#define BATCH   16
#define NPTS    2500
#define NPAD    2560          // multiple of 64 (block n-tile)

// CDNA5 async global->LDS path (ASYNCcnt) if the toolchain exposes it
#if __has_builtin(__builtin_amdgcn_global_load_async_to_lds_b128) && \
    __has_builtin(__builtin_amdgcn_s_wait_asynccnt)
#define HAVE_ASYNC_LDS 1
#else
#define HAVE_ASYNC_LDS 0
#endif

#if HAVE_ASYNC_LDS
__device__ __forceinline__ void async_copy_b128(const bf16_t* g, bf16_t* l)
{
    // Toolchain signature: (v4i AS1*, v4i AS3*, imm offset, imm cpol)
    __builtin_amdgcn_global_load_async_to_lds_b128(
        (__attribute__((address_space(1))) v4i_t*)(void*)(g),
        (__attribute__((address_space(3))) v4i_t*)(void*)(l),
        0, 0);
}
#endif

// ---------------------------------------------------------------------------
// Kernel 1: fold BN into weights, cast to bf16, zero-pad to (Cop x Kp)
// ---------------------------------------------------------------------------
__global__ void fold_weights_kernel(const float* __restrict__ W,
                                    const float* __restrict__ cb,
                                    const float* __restrict__ gamma,
                                    const float* __restrict__ beta,
                                    const float* __restrict__ mean,
                                    const float* __restrict__ var,
                                    bf16_t* __restrict__ Wf,
                                    float* __restrict__ bf,
                                    int Ci, int Co, int Kp, int Cop, int hasBN)
{
    long idx = (long)blockIdx.x * blockDim.x + threadIdx.x;
    long total = (long)Cop * Kp;
    if (idx >= total) return;
    int k = (int)(idx % Kp);
    int o = (int)(idx / Kp);
    float inv = 1.0f, off = 0.0f;
    if (o < Co) {
        float c = cb[o];
        if (hasBN) {
            inv = gamma[o] * rsqrtf(var[o] + 1e-5f);
            off = c * inv + beta[o] - mean[o] * inv;
        } else {
            off = c;
        }
    }
    float w = (o < Co && k < Ci) ? W[(long)o * Ci + k] * inv : 0.0f;
    Wf[(long)o * Kp + k] = (bf16_t)w;
    if (k == 0) bf[o] = (o < Co) ? off : 0.0f;
}

// ---------------------------------------------------------------------------
// Kernel 2: build X0 = concat(output_points, latent, image) in bf16,
// layout (B, NPAD, Kp) point-major / channel-contiguous, zero padded.
// ---------------------------------------------------------------------------
__global__ void build_x0_kernel(const float* __restrict__ op,    // (B,3,N)
                                const float* __restrict__ lat,   // (B,1024)
                                const float* __restrict__ img,   // (B,1024)
                                bf16_t* __restrict__ X0, int Kp)
{
    long idx = (long)blockIdx.x * blockDim.x + threadIdx.x;
    long total = (long)BATCH * NPAD * Kp;
    if (idx >= total) return;
    int c = (int)(idx % Kp);
    long r = idx / Kp;
    int n = (int)(r % NPAD);
    int b = (int)(r / NPAD);
    float v = 0.0f;
    if (n < NPTS) {
        if (c < 3)          v = op[((long)b * 3 + c) * NPTS + n];
        else if (c < 1027)  v = lat[(long)b * 1024 + (c - 3)];
        else if (c < 2051)  v = img[(long)b * 1024 + (c - 1027)];
    }
    X0[idx] = (bf16_t)v;
}

// ---------------------------------------------------------------------------
// Kernel 3: WMMA bf16 GEMM + folded-BN bias + ReLU.
//   Y[b,n,co] = relu( sum_k Wf[co,k]*X[b,n,k] + bias[co] )
//
// Block = 128 threads = 4 wave32. Block tile = (16*MT co) x (64 n).
// A (weight) tile per 32-wide K step is shared by all 4 waves, staged in
// LDS via double-buffered GLOBAL_LOAD_ASYNC_TO_LDS_B128 (ASYNCcnt).
// B fragment streams from global, register double-buffered.
// A fragments rotate through two register sets so the ds_loads for sub-tile
// t+1 are in flight while the WMMA of sub-tile t executes.
// ---------------------------------------------------------------------------
template <int MT, bool FINAL>
__global__ void gemm_wmma_kernel(const bf16_t* __restrict__ W,
                                 const float*  __restrict__ bias,
                                 const bf16_t* __restrict__ X,
                                 bf16_t* __restrict__ Y,
                                 float*  __restrict__ finalOut,
                                 int Kp, int Cop)
{
    const int lane = threadIdx.x & 31;
    const int wave = threadIdx.x >> 5;
    const int hi   = lane >> 4;     // 0/1: which K half this lane serves
    const int l16  = lane & 15;

    const int coBase = blockIdx.x * (16 * MT);
    const int nBase  = blockIdx.y * 64 + wave * 16;
    const int b      = blockIdx.z;

    const bf16_t* __restrict__ xrow = X + ((long)b * NPAD + nBase + l16) * Kp;

    v8f acc[MT];
#pragma unroll
    for (int t = 0; t < MT; ++t) acc[t] = (v8f)(0.0f);

    const int nsteps = Kp >> 5;

#if HAVE_ASYNC_LDS
    __shared__ __align__(16) bf16_t tileA[2][MT * 16][40];  // 32 K + 8 pad
    constexpr int PER = (MT * 64 + 127) / 128;   // async b128 per thread/tile

    // Per-thread copy job: one (MT=1: threads 0..63) or two (MT=4) b128
    // chunks per tile. Pointers precomputed; only the uniform k advances.
    const int crow  = threadIdx.x >> 2;                    // 0..31
    const int crowc = (MT * 16 > 32) ? crow : (crow & 15); // clamp for MT=1
    const int ccol  = (threadIdx.x & 3) * 8;               // bf16 elems (16B)
    const bf16_t* gsrc0 = W + (long)(coBase + crowc) * Kp + ccol;
    const bf16_t* gsrc1 = (MT * 16 > 32)
                        ? W + (long)(coBase + crow + 32) * Kp + ccol : gsrc0;
    bf16_t* l0[2] = { &tileA[0][crowc][ccol], &tileA[1][crowc][ccol] };
    bf16_t* l1[2] = { &tileA[0][(MT * 16 > 32) ? crow + 32 : crowc][ccol],
                      &tileA[1][(MT * 16 > 32) ? crow + 32 : crowc][ccol] };

    auto issue_copy = [&](int buf, int k) {
        if (MT * 16 > 32) {                  // MT=4: all 128 threads, 2 chunks
            async_copy_b128(gsrc0 + k, l0[buf]);
            async_copy_b128(gsrc1 + k, l1[buf]);
        } else {                             // MT=1: threads 0..63, 1 chunk
            if (threadIdx.x < MT * 64) async_copy_b128(gsrc0 + k, l0[buf]);
        }
    };

    // Prologue: tile 0 in flight, B fragment 0 in registers.
    issue_copy(0, 0);
    FragBF16 bcur;
    {
        const v4u* bp = (const v4u*)(xrow + 8 * hi);
        bcur.u[0] = bp[0];
        bcur.u[1] = bp[2];
    }

    for (int s = 0; s < nsteps; ++s) {
        __syncthreads();   // everyone finished reading buffer (s+1)&1
        const int kn = ((s + 1 < nsteps) ? s + 1 : s) * 32;   // clamped next k
        issue_copy((s + 1) & 1, kn);
        __builtin_amdgcn_s_wait_asynccnt(PER);   // tile s complete (in-order)
        __syncthreads();   // tile s visible to all waves

        // Register double-buffer of the streamed B fragment (clamped, no branch)
        FragBF16 bnext;
        {
            const v4u* bp = (const v4u*)(xrow + kn + 8 * hi);
            bnext.u[0] = bp[0];
            bnext.u[1] = bp[2];
        }
        __builtin_prefetch(xrow + ((s + 8 < nsteps) ? s + 8 : s) * 32, 0, 1);

        // Rotate A fragments: ds_loads for t+1 issued before WMMA of t.
        FragBF16 afc, afn;
        {
            const v4u* lp = (const v4u*)(&tileA[s & 1][l16][0]);
            afc.u[0] = lp[hi];        // K = [8*hi,    +8)
            afc.u[1] = lp[2 + hi];    // K = [16+8*hi, +8)
        }
#pragma unroll
        for (int t = 0; t < MT; ++t) {
            if (t + 1 < MT) {
                const v4u* lp = (const v4u*)(&tileA[s & 1][16 * (t + 1) + l16][0]);
                afn.u[0] = lp[hi];
                afn.u[1] = lp[2 + hi];
            }
            acc[t] = __builtin_amdgcn_wmma_f32_16x16x32_bf16(
                false, afc.v, false, bcur.v, (short)0, acc[t], false, false);
            if (t + 1 < MT) afc = afn;
        }
        bcur = bnext;
    }
#else
    // Fallback: synchronous reg->LDS staging, single buffer.
    __shared__ __align__(16) bf16_t tileA[MT * 16][40];
    for (int s = 0; s < nsteps; ++s) {
        const int k = s * 32;
        __syncthreads();
        for (int c = threadIdx.x; c < MT * 64; c += 128) {
            int row = c >> 2;
            int col = (c & 3) * 8;
            *(v4u*)&tileA[row][col] =
                *(const v4u*)(W + (long)(coBase + row) * Kp + k + col);
        }
        __syncthreads();

        FragBF16 bf_;
        const v4u* bp = (const v4u*)(xrow + k + 8 * hi);
        bf_.u[0] = bp[0];
        bf_.u[1] = bp[2];

        FragBF16 afc, afn;
        {
            const v4u* lp = (const v4u*)(&tileA[l16][0]);
            afc.u[0] = lp[hi];
            afc.u[1] = lp[2 + hi];
        }
#pragma unroll
        for (int t = 0; t < MT; ++t) {
            if (t + 1 < MT) {
                const v4u* lp = (const v4u*)(&tileA[16 * (t + 1) + l16][0]);
                afn.u[0] = lp[hi];
                afn.u[1] = lp[2 + hi];
            }
            acc[t] = __builtin_amdgcn_wmma_f32_16x16x32_bf16(
                false, afc.v, false, bf_.v, (short)0, acc[t], false, false);
            if (t + 1 < MT) afc = afn;
        }
    }
#endif

    if (!FINAL) {
#pragma unroll
        for (int t = 0; t < MT; ++t) {
            const int co0 = coBase + 16 * t + 8 * hi;  // 8 consecutive channels
            PackBF8 o;
#pragma unroll
            for (int v = 0; v < 8; ++v) {
                float r = acc[t][v] + bias[co0 + v];
                r = fmaxf(r, 0.0f);
                o.v[v] = (bf16_t)r;
            }
            *(v4u*)(Y + ((long)b * NPAD + nBase + l16) * Cop + co0) = o.u;
        }
    } else {
        const int n = nBase + l16;
        if (n < NPTS) {
            const int co0 = 8 * hi;
#pragma unroll
            for (int v = 0; v < 8; ++v) {
                int co = co0 + v;
                if (co < 3)
                    finalOut[((long)b * NPTS + n) * 3 + co] = acc[0][v] + bias[co];
            }
        }
    }
}

// ---------------------------------------------------------------------------
// Kernel 4: brute-force NN (argmin over out_pts) + color gather
// ---------------------------------------------------------------------------
#define NNTS 256
__global__ void nn_gather_kernel(const float* __restrict__ points,  // (B,M,6)
                                 const float* __restrict__ op,      // (B,3,N)
                                 float* __restrict__ out)           // (B,M,3)
{
    __shared__ float sx[NNTS], sy[NNTS], sz[NNTS];
    const int b = blockIdx.y;
    const int m = blockIdx.x * blockDim.x + threadIdx.x;
    const bool valid = (m < NPTS);

    float gx = 0.f, gy = 0.f, gz = 0.f;
    if (valid) {
        const float* p = points + ((long)b * NPTS + m) * 6;
        gx = p[0]; gy = p[1]; gz = p[2];
    }
    float best = 3.4e38f;
    int bidx = 0;
    const float* opb = op + (long)b * 3 * NPTS;

    for (int t0 = 0; t0 < NPTS; t0 += NNTS) {
        int j = t0 + threadIdx.x;
        __syncthreads();
        if (j < NPTS) {
            sx[threadIdx.x] = opb[j];
            sy[threadIdx.x] = opb[NPTS + j];
            sz[threadIdx.x] = opb[2 * NPTS + j];
        }
        __syncthreads();
        int lim = min(NNTS, NPTS - t0);
#pragma unroll 4
        for (int jj = 0; jj < lim; ++jj) {
            float dx = gx - sx[jj], dy = gy - sy[jj], dz = gz - sz[jj];
            float d = dx * dx + dy * dy + dz * dz;
            if (d < best) { best = d; bidx = t0 + jj; }
        }
    }
    if (valid) {
        const float* pc = points + ((long)b * NPTS + bidx) * 6 + 3;
        float* o = out + ((long)b * NPTS + m) * 3;
        o[0] = pc[0]; o[1] = pc[1]; o[2] = pc[2];
    }
}

// ---------------------------------------------------------------------------
// Host launcher
// ---------------------------------------------------------------------------
static inline size_t align256(size_t x) { return (x + 255) & ~(size_t)255; }

extern "C" void kernel_launch(void* const* d_in, const int* in_sizes, int n_in,
                              void* d_out, int out_size, void* d_ws, size_t ws_size,
                              hipStream_t stream)
{
    (void)in_sizes; (void)n_in; (void)out_size; (void)ws_size;

    // Input order: op, lat, img, points, conv_w[0..6], conv_b[0..6],
    //              gamma[0..5], beta[0..5], mean[0..5], var[0..5]
    const float* op     = (const float*)d_in[0];
    const float* lat    = (const float*)d_in[1];
    const float* img    = (const float*)d_in[2];
    const float* points = (const float*)d_in[3];
    const float* conv_w[7]; const float* conv_b[7];
    for (int i = 0; i < 7; ++i) { conv_w[i] = (const float*)d_in[4 + i];
                                  conv_b[i] = (const float*)d_in[11 + i]; }
    const float* gm[6]; const float* bt[6]; const float* mn[6]; const float* vr[6];
    for (int i = 0; i < 6; ++i) { gm[i] = (const float*)d_in[18 + i];
                                  bt[i] = (const float*)d_in[24 + i];
                                  mn[i] = (const float*)d_in[30 + i];
                                  vr[i] = (const float*)d_in[36 + i]; }

    static const int Cp[8] = {2080, 2112, 1024, 512, 256, 128, 64, 16};
    static const int Ci[7] = {2051, 2096, 1024, 512, 256, 128, 64};
    static const int Co[7] = {2096, 1024, 512, 256, 128, 64, 3};

    // Workspace layout: [bufA][bufB][Wf (bf16)][bias (f32)]
    size_t szA = align256((size_t)BATCH * NPAD * 2080 * sizeof(bf16_t));
    size_t szB = align256((size_t)BATCH * NPAD * 2112 * sizeof(bf16_t));
    size_t wfOff[8]; wfOff[0] = 0;
    size_t bOff[8];  bOff[0]  = 0;
    for (int i = 0; i < 7; ++i) {
        wfOff[i + 1] = wfOff[i] + (size_t)Cp[i + 1] * Cp[i];
        bOff[i + 1]  = bOff[i]  + (size_t)Cp[i + 1];
    }
    char* base = (char*)d_ws;
    bf16_t* bufA  = (bf16_t*)base;
    bf16_t* bufB  = (bf16_t*)(base + szA);
    bf16_t* wfAll = (bf16_t*)(base + szA + szB);
    float*  bAll  = (float*)(base + szA + szB + align256(wfOff[7] * sizeof(bf16_t)));

    // 1) Fold BN into weights (bf16, padded)
    for (int i = 0; i < 7; ++i) {
        long total = (long)Cp[i + 1] * Cp[i];
        int blocks = (int)((total + 255) / 256);
        int hasBN = (i < 6) ? 1 : 0;
        const float* g  = hasBN ? gm[i] : gm[0];
        const float* be = hasBN ? bt[i] : bt[0];
        const float* me = hasBN ? mn[i] : mn[0];
        const float* va = hasBN ? vr[i] : vr[0];
        fold_weights_kernel<<<blocks, 256, 0, stream>>>(
            conv_w[i], conv_b[i], g, be, me, va,
            wfAll + wfOff[i], bAll + bOff[i],
            Ci[i], Co[i], Cp[i], Cp[i + 1], hasBN);
    }

    // 2) Build X0 into bufA
    {
        long total = (long)BATCH * NPAD * Cp[0];
        int blocks = (int)((total + 255) / 256);
        build_x0_kernel<<<blocks, 256, 0, stream>>>(op, lat, img, bufA, Cp[0]);
    }

    // 3) 6 hidden GEMM+BN+ReLU layers, ping-pong bufA <-> bufB
    bf16_t* cur = bufA;
    bf16_t* nxt = bufB;
    dim3 block(128);  // 4 wave32
    for (int i = 0; i < 6; ++i) {
        dim3 grid(Cp[i + 1] / 64, NPAD / 64, BATCH);
        gemm_wmma_kernel<4, false><<<grid, block, 0, stream>>>(
            wfAll + wfOff[i], bAll + bOff[i], cur, nxt, nullptr, Cp[i], Cp[i + 1]);
        bf16_t* t = cur; cur = nxt; nxt = t;
    }

    // 4) Final layer -> pred_colors (f32) at d_out[0 .. B*N*3)
    {
        dim3 grid(1, NPAD / 64, BATCH);
        gemm_wmma_kernel<1, true><<<grid, block, 0, stream>>>(
            wfAll + wfOff[6], bAll + bOff[6], cur, nullptr, (float*)d_out,
            Cp[6], Cp[7]);
    }

    // 5) NN search + color gather -> gt_colors at d_out[B*N*3 ..)
    {
        dim3 grid((NPTS + NNTS - 1) / NNTS, BATCH);
        nn_gather_kernel<<<grid, NNTS, 0, stream>>>(
            points, op, (float*)d_out + (size_t)BATCH * NPTS * 3);
    }
}
